// GCN_72095321030789
// MI455X (gfx1250) — matrix operand order repaired and checked by reading, not compiled, not run on hardware
//
#include <hip/hip_runtime.h>
#include <hip/hip_bf16.h>

// ---------------------------------------------------------------------------
// GCN (3x GCNConv + log_softmax) for MI455X / gfx1250, wave32 + WMMA bf16.
// GEMMs: v_wmma_f32_16x16x32_bf16, weights staged in LDS (bank-conflict-free
// padded stride), A fragments prefetched in one clause, B fragments double-
// buffered across K-chunks so the 8 independent WMMAs per chunk issue
// back-to-back. Edge aggregation: L2-resident float4 gathers + global f32
// atomics with norm = dinv[src]*dinv[dst].
// ---------------------------------------------------------------------------

typedef __bf16  bf16_t;
typedef bf16_t  v16bf __attribute__((ext_vector_type(16)));
typedef bf16_t  v8bf  __attribute__((ext_vector_type(8)));
typedef float   v8f   __attribute__((ext_vector_type(8)));

__device__ __forceinline__ bf16_t f2bf(float f) {
    // round-to-nearest-even f32 -> bf16
    unsigned u = __float_as_uint(f);
    unsigned r = (u + 0x7FFFu + ((u >> 16) & 1u)) >> 16;
    unsigned short s = (unsigned short)r;
    bf16_t b;
    __builtin_memcpy(&b, &s, sizeof(s));
    return b;
}

// ---------------- small prep kernels ----------------

__global__ __launch_bounds__(256) void k_f32_to_bf16(const float* __restrict__ in,
                                                     bf16_t* __restrict__ out,
                                                     long long n) {
    long long i = (long long)blockIdx.x * blockDim.x + threadIdx.x;
    if (i < n) out[i] = f2bf(in[i]);
}

// Wt[n*K + k] = W[k*NOUT + n]  (bf16, K-contiguous columns for B fragments)
__global__ __launch_bounds__(256) void k_transpose_w(const float* __restrict__ W,
                                                     bf16_t* __restrict__ Wt,
                                                     int K, int NOUT) {
    int i = blockIdx.x * blockDim.x + threadIdx.x;
    if (i < K * NOUT) {
        int n = i / K, k = i - n * K;
        Wt[i] = f2bf(W[k * NOUT + n]);
    }
}

__global__ __launch_bounds__(256) void k_fill(float* __restrict__ p, float v, long long n) {
    long long i = (long long)blockIdx.x * blockDim.x + threadIdx.x;
    if (i < n) p[i] = v;
}

__global__ __launch_bounds__(256) void k_deg(const int* __restrict__ dst,
                                             float* __restrict__ deg, int E) {
    int e = blockIdx.x * blockDim.x + threadIdx.x;
    if (e < E) atomicAdd(&deg[dst[e]], 1.0f);
}

__global__ __launch_bounds__(256) void k_rsqrt(float* __restrict__ p, int n) {
    int i = blockIdx.x * blockDim.x + threadIdx.x;
    if (i < n) {
        float d = p[i];
        p[i] = d > 0.0f ? rsqrtf(d) : 0.0f;
    }
}

// ---------------- WMMA GEMM: C[M x NOUT] = A[M x 128] @ W ----------------
// A: bf16 row-major (K=128 contiguous). Wt: bf16 [NOUT][128] (K contiguous),
// staged in LDS with padded stride 136 (272B) so column reads hit distinct
// banks. One wave per 16-row strip; all 4 A K-chunks prefetched; B fragments
// double-buffered across K-chunks -> NT independent WMMAs issue in a row
// while the next chunk's ds_load_b128s are in flight.

template <int NOUT>
__global__ __launch_bounds__(256, 2) void k_gemm_wmma(const bf16_t* __restrict__ A,
                                                      const bf16_t* __restrict__ Wt,
                                                      float* __restrict__ C, int M) {
    constexpr int NT = NOUT / 16;
    constexpr int LSTR = 136;                         // padded LDS row stride (bf16)
    __shared__ bf16_t lw[NOUT * LSTR];

    const int lane  = threadIdx.x & 31;
    const int wave  = threadIdx.x >> 5;
    const int strip = blockIdx.x * 8 + wave;          // wave-uniform

    // cooperative stage of Wt into LDS (16B chunks), before any wave exits
    #pragma unroll
    for (int c = threadIdx.x; c < NOUT * 16; c += 256) {
        int col = c >> 4, part = c & 15;              // 16 chunks of 8 bf16 per column
        *(float4*)(&lw[col * LSTR + part * 8]) = ((const float4*)Wt)[c];
    }
    __syncthreads();
    if (strip * 16 >= M) return;                      // whole-wave exit (EXEC all-1s)

    const int row    = strip * 16 + (lane & 15);      // A-fragment row (M = lane&15)
    const int kbaseA = (lane >> 4) * 8;               // lanes 16-31 hold K+8
    const int kbaseB = (lane >> 4) * 8;               // lanes 16-31 hold K+8
    const int colB   = lane & 15;                     // B-fragment column

    // prefetch all four A K-chunks (8 global_load_b128 in one clause)
    const bf16_t* arow = A + (size_t)row * 128;
    v16bf afrag[4];
    #pragma unroll
    for (int kc = 0; kc < 4; ++kc) {
        const bf16_t* ap = arow + kc * 32 + kbaseA;
        v8bf alo = *(const v8bf*)(ap);                // K = kc*32+kbaseA .. +7
        v8bf ahi = *(const v8bf*)(ap + 16);           // K = kc*32+16+kbaseA .. +7
        afrag[kc] = __builtin_shufflevector(alo, ahi,
            0, 1, 2, 3, 4, 5, 6, 7, 8, 9, 10, 11, 12, 13, 14, 15);
    }

    // double-buffered B fragments from LDS
    v16bf bfrag[2][NT];
    auto loadB = [&](int kc, int buf) {
        #pragma unroll
        for (int nt = 0; nt < NT; ++nt) {
            const bf16_t* bp = &lw[(nt * 16 + colB) * LSTR + kc * 32 + kbaseB];
            v8bf blo = *(const v8bf*)(bp);
            v8bf bhi = *(const v8bf*)(bp + 16);
            bfrag[buf][nt] = __builtin_shufflevector(blo, bhi,
                0, 1, 2, 3, 4, 5, 6, 7, 8, 9, 10, 11, 12, 13, 14, 15);
        }
    };

    loadB(0, 0);
    v8f acc[NT] = {};
    #pragma unroll
    for (int kc = 0; kc < 4; ++kc) {
        if (kc < 3) loadB(kc + 1, (kc + 1) & 1);      // prefetch next chunk's B
        #pragma unroll
        for (int nt = 0; nt < NT; ++nt)
            acc[nt] = __builtin_amdgcn_wmma_f32_16x16x32_bf16(
                false, afrag[kc], false, bfrag[kc & 1][nt], (short)0,
                acc[nt], false, false);
    }

    // C/D layout: VGPR v -> row = v + 8*(lane>=16), col = lane&15
    const int rtop = strip * 16 + (lane >> 4) * 8;
    #pragma unroll
    for (int nt = 0; nt < NT; ++nt)
        #pragma unroll
        for (int v = 0; v < 8; ++v)
            C[(size_t)(rtop + v) * NOUT + nt * 16 + colB] = acc[nt][v];
}

// ---------------- edge aggregation: out[dst] += norm * t[src] ----------------
// One wave per message (edges, then N self-loops). L2-resident gathers+atomics.

template <int D>
__global__ __launch_bounds__(256) void k_aggregate(const float* __restrict__ t,
                                                   const int* __restrict__ src,
                                                   const int* __restrict__ dst,
                                                   const float* __restrict__ dinv,
                                                   float* __restrict__ out,
                                                   int E, int N) {
    constexpr int VW = D / 32;
    typedef float fvec __attribute__((ext_vector_type(VW)));
    long long id = ((long long)blockIdx.x * blockDim.x + threadIdx.x) >> 5;
    const int lane = threadIdx.x & 31;
    if (id >= (long long)E + N) return;               // wave-uniform

    int s, d; float nrm;
    if (id < E) {
        s = src[id]; d = dst[id];
        nrm = dinv[s] * dinv[d];
    } else {
        int i = (int)(id - E);
        s = i; d = i;
        float di = dinv[i];
        nrm = di * di;                                 // self-loop
    }
    const fvec v = *(const fvec*)(t + (size_t)s * D + lane * VW);
    float* orow = out + (size_t)d * D + lane * VW;
    #pragma unroll
    for (int j = 0; j < VW; ++j) atomicAdd(orow + j, v[j] * nrm);
}

// ---------------- epilogue: bf16(relu(agg + bias)) ----------------

__global__ __launch_bounds__(256) void k_bias_relu_bf16(const float* __restrict__ agg,
                                                        const float* __restrict__ bias,
                                                        bf16_t* __restrict__ out,
                                                        long long n, int colmask) {
    long long i = (long long)blockIdx.x * blockDim.x + threadIdx.x;
    if (i < n) {
        float v = agg[i] + bias[(int)i & colmask];
        out[i] = f2bf(fmaxf(v, 0.0f));
    }
}

// ---------------- log_softmax over 64 cols: one wave per node ----------------

__global__ __launch_bounds__(256) void k_logsoftmax64(const float* __restrict__ agg,
                                                      const float* __restrict__ b3,
                                                      float* __restrict__ out, int N) {
    long long wid = ((long long)blockIdx.x * blockDim.x + threadIdx.x) >> 5;
    const int lane = threadIdx.x & 31;
    if (wid >= N) return;
    const float* r = agg + (size_t)wid * 64;
    float v0 = r[lane]      + b3[lane];
    float v1 = r[lane + 32] + b3[lane + 32];
    float m = fmaxf(v0, v1);
    #pragma unroll
    for (int o = 16; o > 0; o >>= 1) m = fmaxf(m, __shfl_xor(m, o, 32));
    float s = expf(v0 - m) + expf(v1 - m);
    #pragma unroll
    for (int o = 16; o > 0; o >>= 1) s += __shfl_xor(s, o, 32);
    float ls = logf(s);
    out[(size_t)wid * 64 + lane]      = v0 - m - ls;
    out[(size_t)wid * 64 + lane + 32] = v1 - m - ls;
}

// ---------------- launcher ----------------

static inline unsigned gsz(long long n, int b) { return (unsigned)((n + b - 1) / b); }

extern "C" void kernel_launch(void* const* d_in, const int* in_sizes, int n_in,
                              void* d_out, int out_size, void* d_ws, size_t ws_size,
                              hipStream_t stream) {
    (void)n_in; (void)out_size; (void)ws_size;
    const float* x  = (const float*)d_in[0];
    const int*   ei = (const int*)d_in[1];
    const float* W1 = (const float*)d_in[2];
    const float* b1 = (const float*)d_in[3];
    const float* W2 = (const float*)d_in[4];
    const float* b2 = (const float*)d_in[5];
    const float* W3 = (const float*)d_in[6];
    const float* b3 = (const float*)d_in[7];

    const long long N = in_sizes[0] / 128;   // 100000
    const int       E = in_sizes[1] / 2;     // 1600000
    const int* src = ei;
    const int* dst = ei + E;

    // carve 256B-aligned scratch
    char* wp = (char*)d_ws;
    auto carve = [&](size_t bytes) -> char* {
        char* p = wp; wp += (bytes + 255) & ~(size_t)255; return p;
    };
    float*  dinv = (float*)carve((size_t)N * 4);
    float*  t    = (float*)carve((size_t)N * 128 * 4);
    float*  agg  = (float*)carve((size_t)N * 128 * 4);
    bf16_t* xbf  = (bf16_t*)carve((size_t)N * 128 * 2);
    bf16_t* hbf  = (bf16_t*)carve((size_t)N * 128 * 2);
    bf16_t* Wt1  = (bf16_t*)carve(128 * 128 * 2);
    bf16_t* Wt2  = (bf16_t*)carve(128 * 128 * 2);
    bf16_t* Wt3  = (bf16_t*)carve(64 * 128 * 2);

    const long long NF = N * 128;
    const long long nMsg = (long long)E + N;
    const int strips  = (int)((N + 15) / 16);
    const int gblocks = (strips + 7) / 8;    // 8 waves per block

    // prep
    k_f32_to_bf16<<<gsz(NF, 256), 256, 0, stream>>>(x, xbf, NF);
    k_transpose_w<<<gsz(128 * 128, 256), 256, 0, stream>>>(W1, Wt1, 128, 128);
    k_transpose_w<<<gsz(128 * 128, 256), 256, 0, stream>>>(W2, Wt2, 128, 128);
    k_transpose_w<<<gsz(64 * 128, 256), 256, 0, stream>>>(W3, Wt3, 128, 64);
    k_fill<<<gsz(N, 256), 256, 0, stream>>>(dinv, 1.0f, N);       // self-loop degree
    k_deg<<<gsz(E, 256), 256, 0, stream>>>(dst, dinv, E);
    k_rsqrt<<<gsz(N, 256), 256, 0, stream>>>(dinv, (int)N);

    // layer 1: t = xbf @ W1 ; agg ; h = relu(agg + b1)
    k_gemm_wmma<128><<<gblocks, 256, 0, stream>>>(xbf, Wt1, t, (int)N);
    k_fill<<<gsz(NF, 256), 256, 0, stream>>>(agg, 0.0f, NF);
    k_aggregate<128><<<gsz(nMsg * 32, 256), 256, 0, stream>>>(t, src, dst, dinv, agg, E, (int)N);
    k_bias_relu_bf16<<<gsz(NF, 256), 256, 0, stream>>>(agg, b1, hbf, NF, 127);

    // layer 2 (reuse xbf as next hidden)
    k_gemm_wmma<128><<<gblocks, 256, 0, stream>>>(hbf, Wt2, t, (int)N);
    k_fill<<<gsz(NF, 256), 256, 0, stream>>>(agg, 0.0f, NF);
    k_aggregate<128><<<gsz(nMsg * 32, 256), 256, 0, stream>>>(t, src, dst, dinv, agg, E, (int)N);
    k_bias_relu_bf16<<<gsz(NF, 256), 256, 0, stream>>>(agg, b2, xbf, NF, 127);

    // layer 3 (NOUT=64) + log_softmax
    k_gemm_wmma<64><<<gblocks, 256, 0, stream>>>(xbf, Wt3, t, (int)N);
    k_fill<<<gsz(N * 64, 256), 256, 0, stream>>>(agg, 0.0f, N * 64);
    k_aggregate<64><<<gsz(nMsg * 32, 256), 256, 0, stream>>>(t, src, dst, dinv, agg, E, (int)N);
    k_logsoftmax64<<<gsz(N * 32, 256), 256, 0, stream>>>(agg, b3, (float*)d_out, (int)N);
}
